// MIM_67997922230511
// MI455X (gfx1250) — compile-verified, matrix operand order checked by reading, not run on hardware
//
#include <hip/hip_runtime.h>

// ---------------------------------------------------------------------------
// Types for CDNA5 WMMA (wave32)
// ---------------------------------------------------------------------------
typedef __bf16 bf16_t;
typedef __bf16 v16bf __attribute__((ext_vector_type(16)));
typedef float  v8f   __attribute__((ext_vector_type(8)));
typedef float  f32x4 __attribute__((ext_vector_type(4)));
typedef unsigned u32x4 __attribute__((ext_vector_type(4)));
typedef unsigned u32x2 __attribute__((ext_vector_type(2)));

#define LDSS 40  // bf16 row stride: 80B rows -> 16B aligned, odd bank phase (20 mod 64)

__device__ __forceinline__ v16bf load_frag(const bf16_t* base, int row0, int lane) {
    // 16-bit A/B fragment layout (ISA 7.12.2):
    // lane = 16*khalf + r ; VGPRs 0-3: k = khalf*8 + 0..7 ; VGPRs 4-7: k = 16 + khalf*8 + 0..7
    // -> two contiguous 16-byte runs per lane: use 2x ds_load_b128.
    int rr = row0 + (lane & 15);
    int kh = (lane >> 4) * 8;
    const bf16_t* p = base + rr * LDSS + kh;
    union { u32x4 u[2]; v16bf v; } f;
    f.u[0] = *(const u32x4*)(p);
    f.u[1] = *(const u32x4*)(p + 16);
    return f.v;
}

__device__ __forceinline__ v8f zero_v8f() {
    v8f z;
#pragma unroll
    for (int i = 0; i < 8; ++i) z[i] = 0.0f;
    return z;
}

__device__ __forceinline__ unsigned pack2(float x, float y) {
    union { bf16_t h[2]; unsigned u; } p;
    p.h[0] = (bf16_t)x; p.h[1] = (bf16_t)y;
    return p.u;
}

__device__ __forceinline__ void store4(bf16_t* dst, f32x4 v) {
    u32x2 pk;
    pk.x = pack2(v.x, v.y);
    pk.y = pack2(v.z, v.w);
    *(u32x2*)dst = pk;  // 8B aligned: 80*r + 8*c4
}

// ---------------------------------------------------------------------------
// Generic NT GEMM:  C[m,n] = sum_k A[m,k] * Bt[n,k]  (+ bias[n]) (+ C if acc)
// fp32 in/out, bf16 WMMA compute, f32 accumulate.
// Requirements: M % 64 == 0, K % 32 == 0 (all call sites satisfy this).
// Block: 256 threads (8 waves). C tile: 64 (M) x 128 (N). K step: 32.
// ---------------------------------------------------------------------------
__global__ __launch_bounds__(256) void gemm_nt_bf16(
    const float* __restrict__ A, const float* __restrict__ Bt,
    float* __restrict__ C, int M, int N, int K,
    int lda, int ldb, int ldc,
    const float* __restrict__ bias, int acc) {
    __shared__ __align__(16) bf16_t As[64 * LDSS];
    __shared__ __align__(16) bf16_t Bs[128 * LDSS];

    const int tid = threadIdx.x;
    const int m0 = blockIdx.x * 64;
    const int n0 = blockIdx.y * 128;
    const int w = tid >> 5, lane = tid & 31;
    const f32x4 zf4 = {0.0f, 0.0f, 0.0f, 0.0f};

    v8f accv[4];
#pragma unroll
    for (int j = 0; j < 4; ++j) accv[j] = zero_v8f();

    for (int k0 = 0; k0 < K; k0 += 32) {
        // batched wide loads into registers (overlaps previous compute)
        f32x4 av[2], bv[4];
#pragma unroll
        for (int i = 0; i < 2; ++i) {  // A tile 64x32 = 512 float4
            int idx = tid + i * 256; int r = idx >> 3, c4 = idx & 7;
            av[i] = *(const f32x4*)(A + (size_t)(m0 + r) * lda + k0 + c4 * 4);
        }
#pragma unroll
        for (int i = 0; i < 4; ++i) {  // B tile 128x32 = 1024 float4
            int idx = tid + i * 256; int r = idx >> 3, c4 = idx & 7;
            int gn = n0 + r;
            bv[i] = (gn < N) ? *(const f32x4*)(Bt + (size_t)gn * ldb + k0 + c4 * 4) : zf4;
        }
        __syncthreads();  // previous K-step consumers done
#pragma unroll
        for (int i = 0; i < 2; ++i) {
            int idx = tid + i * 256; int r = idx >> 3, c4 = idx & 7;
            store4(&As[r * LDSS + c4 * 4], av[i]);
        }
#pragma unroll
        for (int i = 0; i < 4; ++i) {
            int idx = tid + i * 256; int r = idx >> 3, c4 = idx & 7;
            store4(&Bs[r * LDSS + c4 * 4], bv[i]);
        }
        __syncthreads();

        v16bf a = load_frag(As, 16 * (w & 3), lane);
#pragma unroll
        for (int j = 0; j < 4; ++j) {
            v16bf b = load_frag(Bs, 64 * (w >> 2) + 16 * j, lane);
            accv[j] = __builtin_amdgcn_wmma_f32_16x16x32_bf16(
                false, a, false, b, (short)0, accv[j], false, false);
        }
    }

    const int nh = lane & 15, mh = lane >> 4;
#pragma unroll
    for (int j = 0; j < 4; ++j) {
        int n = n0 + 64 * (w >> 2) + 16 * j + nh;
        if (n >= N) continue;
        float bv = bias ? bias[n] : 0.0f;
#pragma unroll
        for (int r = 0; r < 8; ++r) {
            int m = m0 + 16 * (w & 3) + 8 * mh + r;
            float val = accv[j][r] + bv;
            if (acc) val += C[(size_t)m * ldc + n];
            C[(size_t)m * ldc + n] = val;
        }
    }
}

// ---------------------------------------------------------------------------
// Final edge GEMM: new_edge[r,n] = edge[r,:].G[n,:] + nboi[(b,l),n] + nboj[(b,m),n] + cb[n]
// r = (b*512 + l)*512 + m ; M = 524288 rows, N = 128, K = 128. Full tiles.
// ---------------------------------------------------------------------------
__global__ __launch_bounds__(256) void edge_out_gemm(
    const float* __restrict__ E, const float* __restrict__ G,
    const float* __restrict__ nboi, const float* __restrict__ nboj,
    const float* __restrict__ cb, float* __restrict__ out) {
    __shared__ __align__(16) bf16_t As[64 * LDSS];
    __shared__ __align__(16) bf16_t Bs[128 * LDSS];

    const int tid = threadIdx.x;
    const size_t m0 = (size_t)blockIdx.x * 64;
    const int w = tid >> 5, lane = tid & 31;

    // pull this block's edge rows toward the caches early (global_prefetch_b8)
    if (tid < 64) __builtin_prefetch(&E[(m0 + tid) * 128], 0, 1);

    v8f accv[4];
#pragma unroll
    for (int j = 0; j < 4; ++j) accv[j] = zero_v8f();

    for (int k0 = 0; k0 < 128; k0 += 32) {
        f32x4 av[2], bv[4];
#pragma unroll
        for (int i = 0; i < 2; ++i) {
            int idx = tid + i * 256; int r = idx >> 3, c4 = idx & 7;
            av[i] = *(const f32x4*)(E + (m0 + r) * 128 + k0 + c4 * 4);
        }
#pragma unroll
        for (int i = 0; i < 4; ++i) {
            int idx = tid + i * 256; int r = idx >> 3, c4 = idx & 7;
            bv[i] = *(const f32x4*)(G + (size_t)r * 128 + k0 + c4 * 4);
        }
        __syncthreads();
#pragma unroll
        for (int i = 0; i < 2; ++i) {
            int idx = tid + i * 256; int r = idx >> 3, c4 = idx & 7;
            store4(&As[r * LDSS + c4 * 4], av[i]);
        }
#pragma unroll
        for (int i = 0; i < 4; ++i) {
            int idx = tid + i * 256; int r = idx >> 3, c4 = idx & 7;
            store4(&Bs[r * LDSS + c4 * 4], bv[i]);
        }
        __syncthreads();

        v16bf a = load_frag(As, 16 * (w & 3), lane);
#pragma unroll
        for (int j = 0; j < 4; ++j) {
            v16bf b = load_frag(Bs, 64 * (w >> 2) + 16 * j, lane);
            accv[j] = __builtin_amdgcn_wmma_f32_16x16x32_bf16(
                false, a, false, b, (short)0, accv[j], false, false);
        }
    }

    const int nh = lane & 15, mh = lane >> 4;
#pragma unroll
    for (int j = 0; j < 4; ++j) {
        int n = 64 * (w >> 2) + 16 * j + nh;
        float cbn = cb[n];
#pragma unroll
        for (int r = 0; r < 8; ++r) {
            size_t row = m0 + 16 * (w & 3) + 8 * mh + r;
            int b = (int)(row >> 18);
            int l = (int)((row >> 9) & 511);
            int m = (int)(row & 511);
            out[row * 128 + n] = accv[j][r]
                + nboi[(size_t)(b * 512 + l) * 128 + n]
                + nboj[(size_t)(b * 512 + m) * 128 + n] + cbn;
        }
    }
}

// ---------------------------------------------------------------------------
// Small helper kernels
// ---------------------------------------------------------------------------

// v_h[h,k] = sum_e a_e[e] * We_w[h*16+e, k];  ch[h] = sum_e a_e[e] * We_b[h*16+e]
__global__ void scorev_kernel(const float* __restrict__ attn,
                              const float* __restrict__ We_w,
                              const float* __restrict__ We_b,
                              float* __restrict__ V, float* __restrict__ ch) {
    int t = blockIdx.x * 256 + threadIdx.x;
    if (t < 1024) {
        int h = t >> 7, k = t & 127;
        float s = 0.0f;
        for (int e = 0; e < 16; ++e) s += attn[128 + e] * We_w[(h * 16 + e) * 128 + k];
        V[t] = s;
    }
    if (t < 8) {
        float s = 0.0f;
        for (int e = 0; e < 16; ++e) s += attn[128 + e] * We_b[t * 16 + e];
        ch[t] = s;
    }
}

// ci = hi + di (row term), cj = hj + dj (col term), per (b,l,h)
__global__ void cicj_kernel(const float* __restrict__ npj, const float* __restrict__ dpj,
                            const float* __restrict__ attn,
                            float* __restrict__ ci, float* __restrict__ cj) {
    int t = blockIdx.x * 256 + threadIdx.x;
    if (t >= 8192) return;
    const float* np = npj + (size_t)t * 64;
    const float* dp = dpj + (size_t)t * 16;
    float hi = 0, hj = 0, di = 0, dj = 0;
    for (int d = 0; d < 64; ++d) { hi += np[d] * attn[d]; hj += np[d] * attn[64 + d]; }
    for (int d = 0; d < 16; ++d) { di += dp[d] * attn[144 + d]; dj += dp[d] * attn[160 + d]; }
    ci[t] = hi + di;
    cj[t] = hj + dj;
}

// Fused score + leaky-relu + diag-mask + softmax. One block per (b,l).
// Edge rows staged with gfx1250 async copies (global_load_async_to_lds_b128).
// probs layout: (B, H, L, L)
__global__ __launch_bounds__(256) void score_softmax_kernel(
    const float* __restrict__ edge, const float* __restrict__ V,
    const float* __restrict__ ch, const float* __restrict__ ci,
    const float* __restrict__ cj, float* __restrict__ probs) {
    const int bl = blockIdx.x;        // b*512 + l
    const int b = bl >> 9, l = bl & 511;
    const int tid = threadIdx.x;

    __shared__ float Vs[8 * 128];
    __shared__ __align__(16) float es[32 * 132];  // padded: stride 132 (bank phase 4)
    __shared__ float ss[512 * 9];                 // padded: stride 9
    __shared__ float chs[8], cis[8];

    for (int i = tid; i < 1024; i += 256) Vs[i] = V[i];
    if (tid < 8) { chs[tid] = ch[tid]; cis[tid] = ci[bl * 8 + tid]; }

    const float* erow = edge + (size_t)bl * 512 * 128;
    const int mi = tid >> 3, h = tid & 7;

    for (int chunk = 0; chunk < 16; ++chunk) {
        int m0 = chunk * 32;
        __syncthreads();  // readers of es from previous chunk done
#pragma unroll
        for (int i = 0; i < 4; ++i) {  // 32x128 f32 tile = 1024 float4, async -> LDS
            int idx = tid + i * 256;
            int r = idx >> 5, c4 = idx & 31;
            const float* g = erow + (size_t)(m0 + r) * 128 + c4 * 4;
            unsigned ldsa = (unsigned)(uintptr_t)(&es[r * 132 + c4 * 4]);
            asm volatile("global_load_async_to_lds_b128 %0, %1, off"
                         :: "v"(ldsa), "v"((unsigned long long)(uintptr_t)g)
                         : "memory");
        }
        asm volatile("s_wait_asynccnt 0x0" ::: "memory");
        __syncthreads();
        int m = m0 + mi;
        float dot = 0.0f;
#pragma unroll 4
        for (int k = 0; k < 128; ++k) dot += es[mi * 132 + k] * Vs[h * 128 + k];
        float s = dot + cis[h] + chs[h] + cj[(b * 512 + m) * 8 + h];
        s = (s > 0.0f) ? s : 0.2f * s;        // leaky relu
        if (m == l) s = -3.0e38f;             // diag mask
        ss[m * 9 + h] = s;
    }
    __syncthreads();

    // softmax over m; wave w handles head h=w (wave32)
    const int w = tid >> 5, lane = tid & 31;
    float vals[16];
    float mx = -3.0e38f;
#pragma unroll
    for (int i = 0; i < 16; ++i) {
        vals[i] = ss[(lane + 32 * i) * 9 + w];
        mx = fmaxf(mx, vals[i]);
    }
#pragma unroll
    for (int o = 16; o >= 1; o >>= 1) mx = fmaxf(mx, __shfl_xor(mx, o, 32));
    float sum = 0.0f;
#pragma unroll
    for (int i = 0; i < 16; ++i) { vals[i] = __expf(vals[i] - mx); sum += vals[i]; }
#pragma unroll
    for (int o = 16; o >= 1; o >>= 1) sum += __shfl_xor(sum, o, 32);
    float inv = 1.0f / sum;
    float* prow = probs + (((size_t)(b * 8 + w) * 512 + l) * 512);
#pragma unroll
    for (int i = 0; i < 16; ++i) prow[lane + 32 * i] = vals[i] * inv;
}

// nwh (B,L,H,DK) -> nwhT (B,H,DK,L)
__global__ void transpose_nwh_kernel(const float* __restrict__ nwh, float* __restrict__ nwhT) {
    int t = blockIdx.x * 256 + threadIdx.x;
    if (t >= 524288) return;
    int d = t & 63, h = (t >> 6) & 7, m = (t >> 9) & 511, b = t >> 18;
    nwhT[(((size_t)(b * 8 + h) * 64 + d) * 512) + m] = nwh[t];
}

// y = npj + lrelu(agg) ; y flat (B,L,DN) with DN index = h*64+d
__global__ void y_kernel(const float* __restrict__ npj, const float* __restrict__ agg,
                         float* __restrict__ y) {
    int t = blockIdx.x * 256 + threadIdx.x;
    if (t >= 524288) return;
    int d = t & 63, h = (t >> 6) & 7, l = (t >> 9) & 511, b = t >> 18;
    float a = agg[(((size_t)(b * 8 + h) * 512 + l) * 64) + d];
    a = (a > 0.0f) ? a : 0.2f * a;
    y[t] = npj[t] + a;
}

// M[hf,k] = sum_e edge_w[f,e] * We2_w[h*16+e, k]
__global__ void mmat_kernel(const float* __restrict__ edge_w, const float* __restrict__ We2_w,
                            float* __restrict__ Mm) {
    int t = blockIdx.x * 256 + threadIdx.x;
    if (t >= 16384) return;
    int hf = t >> 7, k = t & 127;
    int f = hf & 15, h = hf >> 4;
    float s = 0.0f;
    for (int e = 0; e < 16; ++e) s += edge_w[f * 176 + e] * We2_w[(h * 16 + e) * 128 + k];
    Mm[t] = s;
}

// G[o,k] = sum_f oute_w[o,f] * M[f,k]
__global__ void gmat_kernel(const float* __restrict__ oute_w, const float* __restrict__ Mm,
                            float* __restrict__ Gm) {
    int t = blockIdx.x * 256 + threadIdx.x;
    if (t >= 16384) return;
    int o = t >> 7, k = t & 127;
    float s = 0.0f;
    for (int f = 0; f < 128; ++f) s += oute_w[o * 128 + f] * Mm[f * 128 + k];
    Gm[t] = s;
}

// cb[o] = sum_hf oute_w[o,hf] * (edge_b[f] + sum_e edge_w[f,e]*We2_b[h*16+e]) + oute_b[o]
__global__ __launch_bounds__(128) void cb_kernel(
    const float* __restrict__ edge_w, const float* __restrict__ We2_b,
    const float* __restrict__ edge_b, const float* __restrict__ oute_w,
    const float* __restrict__ oute_b, float* __restrict__ cb) {
    __shared__ float bf[128];
    int o = threadIdx.x;
    int f = o & 15, h = o >> 4;
    float s = edge_b[f];
    for (int e = 0; e < 16; ++e) s += edge_w[f * 176 + e] * We2_b[h * 16 + e];
    bf[o] = s;
    __syncthreads();
    float c = oute_b[o];
    for (int j = 0; j < 128; ++j) c += oute_w[o * 128 + j] * bf[j];
    cb[o] = c;
}

// out[o, h*dim + d] = sum_f oute_w[o, h*16+f] * edge_w[f, colbase + d]
__global__ void combad_kernel(const float* __restrict__ oute_w, const float* __restrict__ edge_w,
                              float* __restrict__ out, int colbase, int dim, int total) {
    int t = blockIdx.x * 256 + threadIdx.x;
    if (t >= total) return;
    int n = 8 * dim;
    int o = t / n, rem = t % n;
    int h = rem / dim, d = rem % dim;
    float s = 0.0f;
    for (int f = 0; f < 16; ++f) s += oute_w[o * 128 + h * 16 + f] * edge_w[f * 176 + colbase + d];
    out[t] = s;
}

// ---------------------------------------------------------------------------
// Host launcher
// ---------------------------------------------------------------------------
extern "C" void kernel_launch(void* const* d_in, const int* in_sizes, int n_in,
                              void* d_out, int out_size, void* d_ws, size_t ws_size,
                              hipStream_t stream) {
    (void)in_sizes; (void)n_in; (void)out_size; (void)ws_size;
    const float* node   = (const float*)d_in[0];
    const float* edge   = (const float*)d_in[1];
    const float* diff   = (const float*)d_in[2];
    const float* Wn_w   = (const float*)d_in[3];
    const float* Wn_b   = (const float*)d_in[4];
    const float* We_w   = (const float*)d_in[5];
    const float* We_b   = (const float*)d_in[6];
    const float* Wd_w   = (const float*)d_in[7];
    const float* Wd_b   = (const float*)d_in[8];
    const float* Wh_w   = (const float*)d_in[9];
    const float* Wh_b   = (const float*)d_in[10];
    const float* attn   = (const float*)d_in[11];
    const float* outn_w = (const float*)d_in[12];
    const float* outn_b = (const float*)d_in[13];
    const float* Wn2_w  = (const float*)d_in[14];
    const float* Wn2_b  = (const float*)d_in[15];
    const float* We2_w  = (const float*)d_in[16];
    const float* We2_b  = (const float*)d_in[17];
    const float* Wd2_w  = (const float*)d_in[18];
    const float* Wd2_b  = (const float*)d_in[19];
    const float* edge_w = (const float*)d_in[20];
    const float* edge_b = (const float*)d_in[21];
    const float* oute_w = (const float*)d_in[22];
    const float* oute_b = (const float*)d_in[23];

    float* out_node = (float*)d_out;                 // (2,512,512)
    float* out_edge = out_node + 524288;             // (2,512,512,128)

    // workspace carve-up (floats)
    float* ws = (float*)d_ws;
    size_t off = 0;
    float* npj  = ws + off; off += 524288;   // (B,L,H,DK)
    float* dpj  = ws + off; off += 131072;   // (B,L,H,DDH)
    float* nwh  = ws + off; off += 524288;   // (B,L,H,DK)
    float* nwhT = ws + off; off += 524288;   // (B,H,DK,L)
    float* ci   = ws + off; off += 8192;
    float* cj   = ws + off; off += 8192;
    float* Vv   = ws + off; off += 1024;
    float* ch   = ws + off; off += 8;
    float* prb  = ws + off; off += 4194304;  // (B,H,L,L)
    float* agg  = ws + off; off += 524288;   // (B,H,L,DK)
    float* yv   = ws + off; off += 524288;   // (B,L,DN)
    float* np2  = ws + off; off += 524288;
    float* dp2  = ws + off; off += 131072;
    float* Mm   = ws + off; off += 16384;
    float* Gm   = ws + off; off += 16384;
    float* cbv  = ws + off; off += 128;
    float* AiF  = ws + off; off += 65536;    // (128, 512)
    float* AjF  = ws + off; off += 65536;
    float* DiF  = ws + off; off += 16384;    // (128, 128)
    float* DjF  = ws + off; off += 16384;
    float* nboi = ws + off; off += 131072;   // (B*L, 128)
    float* nboj = ws + off; off += 131072;

    dim3 blk(256);

    // Phase-1 projections
    gemm_nt_bf16<<<dim3(16, 4), blk, 0, stream>>>(node, Wn_w, npj, 1024, 512, 512, 512, 512, 512, Wn_b, 0);
    gemm_nt_bf16<<<dim3(16, 1), blk, 0, stream>>>(diff, Wd_w, dpj, 1024, 128, 128, 128, 128, 128, Wd_b, 0);
    gemm_nt_bf16<<<dim3(128, 1), blk, 0, stream>>>(npj, Wh_w, nwh, 8192, 64, 64, 64, 64, 64, Wh_b, 0);

    // Attention score ingredients
    scorev_kernel<<<4, blk, 0, stream>>>(attn, We_w, We_b, Vv, ch);
    cicj_kernel<<<32, blk, 0, stream>>>(npj, dpj, attn, ci, cj);

    // Fused scores + softmax (streams edge tensor once via async-to-LDS)
    score_softmax_kernel<<<1024, blk, 0, stream>>>(edge, Vv, ch, ci, cj, prb);

    // agg = probs @ node_wh (batched over (b,h) via 16 WMMA GEMM launches)
    transpose_nwh_kernel<<<2048, blk, 0, stream>>>(nwh, nwhT);
    for (int z = 0; z < 16; ++z) {
        gemm_nt_bf16<<<dim3(8, 1), blk, 0, stream>>>(
            prb + (size_t)z * 262144, nwhT + (size_t)z * 32768, agg + (size_t)z * 32768,
            512, 64, 512, 512, 512, 64, nullptr, 0);
    }
    y_kernel<<<2048, blk, 0, stream>>>(npj, agg, yv);

    // new_node (output 0)
    gemm_nt_bf16<<<dim3(16, 4), blk, 0, stream>>>(yv, outn_w, out_node, 1024, 512, 512, 512, 512, 512, outn_b, 0);

    // Phase-2 projections
    gemm_nt_bf16<<<dim3(16, 4), blk, 0, stream>>>(out_node, Wn2_w, np2, 1024, 512, 512, 512, 512, 512, Wn2_b, 0);
    gemm_nt_bf16<<<dim3(16, 1), blk, 0, stream>>>(diff, Wd2_w, dp2, 1024, 128, 128, 128, 128, 128, Wd2_b, 0);

    // Composed weights for the fused edge output
    mmat_kernel<<<64, blk, 0, stream>>>(edge_w, We2_w, Mm);
    gmat_kernel<<<64, blk, 0, stream>>>(oute_w, Mm, Gm);
    cb_kernel<<<1, 128, 0, stream>>>(edge_w, We2_b, edge_b, oute_w, oute_b, cbv);
    combad_kernel<<<256, blk, 0, stream>>>(oute_w, edge_w, AiF, 16, 64, 65536);
    combad_kernel<<<256, blk, 0, stream>>>(oute_w, edge_w, AjF, 80, 64, 65536);
    combad_kernel<<<64,  blk, 0, stream>>>(oute_w, edge_w, DiF, 144, 16, 16384);
    combad_kernel<<<64,  blk, 0, stream>>>(oute_w, edge_w, DjF, 160, 16, 16384);

    // nboi / nboj (second call accumulates into C)
    gemm_nt_bf16<<<dim3(16, 1), blk, 0, stream>>>(np2, AiF, nboi, 1024, 128, 512, 512, 512, 128, nullptr, 0);
    gemm_nt_bf16<<<dim3(16, 1), blk, 0, stream>>>(dp2, DiF, nboi, 1024, 128, 128, 128, 128, 128, nullptr, 1);
    gemm_nt_bf16<<<dim3(16, 1), blk, 0, stream>>>(np2, AjF, nboj, 1024, 128, 512, 512, 512, 128, nullptr, 0);
    gemm_nt_bf16<<<dim3(16, 1), blk, 0, stream>>>(dp2, DjF, nboj, 1024, 128, 128, 128, 128, 128, nullptr, 1);

    // Final fused edge GEMM -> new_edge (output 1)
    edge_out_gemm<<<8192, blk, 0, stream>>>(edge, Gm, nboi, nboj, cbv, out_edge);
}
